// GATv2_35622458753877
// MI455X (gfx1250) — matrix operand order repaired and checked by reading, not compiled, not run on hardware
//
#include <hip/hip_runtime.h>

#define NODES 32768
#define EDGES 262144
#define HEADS 8
#define CHAN  64
#define HC    512   // HEADS*CHAN

#define KSTEP  32
#define BPITCH 40   // halves per LDS row: 32 + 8 pad -> 80B pitch (16B aligned, conflict-free)

typedef _Float16 v8h  __attribute__((ext_vector_type(8)));
typedef _Float16 v16h __attribute__((ext_vector_type(16)));
typedef float    v8f  __attribute__((ext_vector_type(8)));
typedef int      v4i  __attribute__((vector_size(4 * sizeof(int))));

typedef __attribute__((address_space(1))) v4i* gv4i_ptr;
typedef __attribute__((address_space(3))) v4i* lv4i_ptr;

#ifdef __has_builtin
#  if __has_builtin(__builtin_amdgcn_global_load_async_to_lds_b128)
#    define HAVE_ASYNC_LDS 1
#  endif
#endif
#ifndef HAVE_ASYNC_LDS
#  define HAVE_ASYNC_LDS 0
#endif

// ---------------- conversion kernels ----------------

// x [N,118] f32 -> xp [N,128] f16 (zero padded K)
__global__ void conv_x_kernel(const float* __restrict__ x, _Float16* __restrict__ xp) {
    int i = blockIdx.x * blockDim.x + threadIdx.x;
    if (i >= NODES * 128) return;
    int row = i >> 7, c = i & 127;
    xp[i] = (c < 118) ? (_Float16)x[row * 118 + c] : (_Float16)0.0f;
}

// Wn [118,512] f32 -> WnT [512 cols][128 K] f16 col-major (zero padded K)
__global__ void conv_wn_kernel(const float* __restrict__ Wn, _Float16* __restrict__ WnT) {
    int i = blockIdx.x * blockDim.x + threadIdx.x;
    if (i >= 512 * 128) return;
    int col = i >> 7, k = i & 127;
    WnT[i] = (k < 118) ? (_Float16)Wn[k * 512 + col] : (_Float16)0.0f;
}

// Wa [1024,512] f32 -> BT [1024 cols][512 K] f16:
// cols 0..511 = Wa_top col j (-> P1), cols 512..1023 = Wa_bot col j-512 (-> P2)
__global__ void conv_wa_kernel(const float* __restrict__ Wa, _Float16* __restrict__ BT) {
    int i = blockIdx.x * blockDim.x + threadIdx.x;
    if (i >= 1024 * 512) return;
    int col = i >> 9, k = i & 511;
    float v = (col < 512) ? Wa[k * 512 + col] : Wa[(k + 512) * 512 + (col - 512)];
    BT[i] = (_Float16)v;
}

// h f32 -> f16
__global__ void conv_h_kernel(const float* __restrict__ h, _Float16* __restrict__ hf) {
    int i = blockIdx.x * blockDim.x + threadIdx.x;
    if (i >= NODES * HC) return;
    hf[i] = (_Float16)h[i];
}

// ---------------- WMMA GEMM ----------------
// C[M,N] f32 = A[M,K] f16 (row-major) @ B (as Bt[N,K] f16) + bias
// 256 threads = 8 waves; wave w -> rows [w*16,+16); block tile 128M x 64N.
// B panel staged in LDS (shared by all 8 waves) via async-to-LDS,
// A fragments register-prefetched one k-step ahead,
// all 4 B fragments preloaded from LDS before the WMMA burst.
__global__ __launch_bounds__(256) void wmma_gemm_kernel(
        const _Float16* __restrict__ A, const _Float16* __restrict__ Bt,
        const float* __restrict__ bias, float* __restrict__ C, int N, int K) {
    __shared__ _Float16 Bs[64 * BPITCH];

    int tid  = threadIdx.x;
    int lane = tid & 31;
    int wave = tid >> 5;
    int m0 = blockIdx.x * 128 + wave * 16;
    int n0 = blockIdx.y * 64;
    int r15  = lane & 15;     // A row within tile / B column within tile
    int rsel = lane >> 4;     // fragment half-select (ISA layout)

    // cooperative B staging: thread copies 8 halves (one b128) per k-step
    int scol  = tid >> 2;         // 0..63
    int skoff = (tid & 3) * 8;    // 0,8,16,24
    const _Float16* bsrc = Bt + (size_t)(n0 + scol) * K + skoff;
    _Float16*       bdst = &Bs[scol * BPITCH + skoff];

    const _Float16* aptr = A + (size_t)(m0 + r15) * K + rsel * 8;
    const _Float16* bfrag0 = &Bs[r15 * BPITCH + rsel * 8];

    v8f acc[4] = {v8f{}, v8f{}, v8f{}, v8f{}};

    int nk = K / KSTEP;

    // preload A fragment for step 0
    v8h al = *(const v8h*)(aptr);
    v8h ah = *(const v8h*)(aptr + 16);
#if !HAVE_ASYNC_LDS
    v8h breg = *(const v8h*)(bsrc);
#endif

    for (int i = 0; i < nk; ++i) {
        __syncthreads();   // all waves done reading Bs from previous step
#if HAVE_ASYNC_LDS
        __builtin_amdgcn_global_load_async_to_lds_b128(
            (gv4i_ptr)(unsigned long long)(bsrc + (size_t)i * KSTEP),
            (lv4i_ptr)(unsigned int)(unsigned long long)bdst,
            0, 0);
#else
        *(v8h*)bdst = breg;
#endif
        // prefetch next-step A (and B bridge) while staging completes
        v8h alN = al, ahN = ah;
        if (i + 1 < nk) {
            const _Float16* ap = aptr + (size_t)(i + 1) * KSTEP;
            alN = *(const v8h*)(ap);
            ahN = *(const v8h*)(ap + 16);
#if !HAVE_ASYNC_LDS
            breg = *(const v8h*)(bsrc + (size_t)(i + 1) * KSTEP);
#endif
            __builtin_prefetch(aptr + (size_t)(i + 2) * KSTEP, 0, 0);
        }
#if HAVE_ASYNC_LDS
        asm volatile("s_wait_asynccnt 0x0" ::: "memory");
#endif
        __syncthreads();   // Bs writes visible to all waves

        // preload ALL B fragments first so ds latency overlaps the WMMA burst
        v16h bf[4];
#pragma unroll
        for (int t = 0; t < 4; ++t) {
            const _Float16* bfp = bfrag0 + t * 16 * BPITCH;
            v8h bl = *(const v8h*)(bfp);
            v8h bh = *(const v8h*)(bfp + 16);
            bf[t] = __builtin_shufflevector(bl, bh, 0,1,2,3,4,5,6,7,8,9,10,11,12,13,14,15);
        }
        v16h a = __builtin_shufflevector(al, ah, 0,1,2,3,4,5,6,7,8,9,10,11,12,13,14,15);
#pragma unroll
        for (int t = 0; t < 4; ++t)
            acc[t] = __builtin_amdgcn_wmma_f32_16x16x32_f16(
                false, a, false, bf[t], (short)0, acc[t], false, false);
        al = alN; ah = ahN;
    }

    int rowbase = rsel * 8;   // C layout: lanes<16 rows 0..7, lanes>=16 rows 8..15
#pragma unroll
    for (int t = 0; t < 4; ++t) {
        int col = n0 + t * 16 + r15;
        float bv = bias ? bias[col] : 0.0f;
#pragma unroll
        for (int r = 0; r < 8; ++r)
            C[(size_t)(m0 + rowbase + r) * N + col] = acc[t][r] + bv;
    }
}

// ---------------- edge-stage kernels ----------------

__device__ inline void atomicMaxFloat(float* addr, float val) {
    // valid for mixed-sign floats given init to a negative sentinel
    if (val >= 0.0f) atomicMax((int*)addr, __float_as_int(val));
    else             atomicMin((unsigned int*)addr, __float_as_uint(val));
}

// zero h_out, set maxs=-1e30, denom=0
__global__ void init_kernel(float* __restrict__ maxs, float* __restrict__ denom,
                            float* __restrict__ hout) {
    int i = blockIdx.x * blockDim.x + threadIdx.x;
    if (i >= NODES * HC) return;
    hout[i] = 0.0f;
    if (i < NODES * HEADS) { maxs[i] = -1e30f; denom[i] = 0.0f; }
}

// score[e,h] = sum_c lrelu(P1[src]+P2[dst]+ba) * attn_w ; atomic max into maxs[dst,h]
__global__ void score_kernel(const float* __restrict__ P, const int* __restrict__ src,
                             const int* __restrict__ dst, const float* __restrict__ ba,
                             const float* __restrict__ aw, float* __restrict__ score,
                             float* __restrict__ maxs) {
    int tid = blockIdx.x * blockDim.x + threadIdx.x;
    if (tid >= EDGES * HEADS) return;
    int e = tid >> 3, h = tid & 7;
    int s = src[e], d = dst[e];
    const float4* p1 = (const float4*)(P + (size_t)s * 1024 + h * 64);
    const float4* p2 = (const float4*)(P + (size_t)d * 1024 + 512 + h * 64);
    const float4* bb = (const float4*)(ba + h * 64);
    const float4* a4 = (const float4*)(aw + h * 64);
    float acc = 0.0f;
#pragma unroll
    for (int c = 0; c < 16; ++c) {
        float4 v1 = p1[c], v2 = p2[c], b = bb[c], w = a4[c];
        float t;
        t = v1.x + v2.x + b.x; t = t > 0.0f ? t : 0.01f * t; acc += t * w.x;
        t = v1.y + v2.y + b.y; t = t > 0.0f ? t : 0.01f * t; acc += t * w.y;
        t = v1.z + v2.z + b.z; t = t > 0.0f ? t : 0.01f * t; acc += t * w.z;
        t = v1.w + v2.w + b.w; t = t > 0.0f ? t : 0.01f * t; acc += t * w.w;
    }
    score[tid] = acc;
    atomicMaxFloat(&maxs[d * 8 + h], acc);
}

// ex = exp(score - max[dst]); score <- ex; denom[dst] += ex
__global__ void expsum_kernel(const int* __restrict__ dst, float* __restrict__ score,
                              const float* __restrict__ maxs, float* __restrict__ denom) {
    int tid = blockIdx.x * blockDim.x + threadIdx.x;
    if (tid >= EDGES * HEADS) return;
    int e = tid >> 3, h = tid & 7;
    int d = dst[e];
    float ex = __expf(score[tid] - maxs[d * 8 + h]);
    score[tid] = ex;
    unsafeAtomicAdd(&denom[d * 8 + h], ex);
}

// h_out[dst] += h_in[src] * (ex / denom[dst])
__global__ void aggregate_kernel(const int* __restrict__ src, const int* __restrict__ dst,
                                 const float* __restrict__ score, const float* __restrict__ denom,
                                 const float* __restrict__ hin, float* __restrict__ hout) {
    int tid = blockIdx.x * blockDim.x + threadIdx.x;
    if (tid >= EDGES * HEADS) return;
    int e = tid >> 3, h = tid & 7;
    int s = src[e], d = dst[e];
    float alpha = score[tid] / denom[d * 8 + h];
    const float* hs = hin + (size_t)s * HC + h * 64;
    float* ho = hout + (size_t)d * HC + h * 64;
#pragma unroll
    for (int c = 0; c < 64; ++c)
        unsafeAtomicAdd(&ho[c], hs[c] * alpha);
}

// ---------------- driver ----------------

extern "C" void kernel_launch(void* const* d_in, const int* in_sizes, int n_in,
                              void* d_out, int out_size, void* d_ws, size_t ws_size,
                              hipStream_t stream) {
    const float* x    = (const float*)d_in[0];
    // d_in[1] edge_feat: unused (reference computes _eh but never uses it)
    const int*   src  = (const int*)d_in[2];
    const int*   dst  = (const int*)d_in[3];
    const float* Wn   = (const float*)d_in[4];
    const float* bn   = (const float*)d_in[5];
    // d_in[6] We, d_in[7] be: unused
    const float* Wa   = (const float*)d_in[8];
    const float* ba   = (const float*)d_in[9];
    const float* attn = (const float*)d_in[10];
    float* out = (float*)d_out;

    char* w = (char*)d_ws;
    size_t off = 0;
    auto alloc = [&](size_t bytes) -> void* {
        void* p = w + off;
        off += (bytes + 255) & ~(size_t)255;
        return p;
    };
    _Float16* WnT  = (_Float16*)alloc((size_t)512 * 128 * 2);
    _Float16* BT   = (_Float16*)alloc((size_t)1024 * 512 * 2);
    float*    maxs = (float*)alloc((size_t)NODES * HEADS * 4);
    float*    den  = (float*)alloc((size_t)NODES * HEADS * 4);
    float*    scr  = (float*)alloc((size_t)EDGES * HEADS * 4);
    _Float16* hf16 = (_Float16*)alloc((size_t)NODES * HC * 2);
    float*    bufA = (float*)alloc((size_t)NODES * HC * 4);
    float*    bufB = (float*)alloc((size_t)NODES * HC * 4);
    float*    P    = (float*)alloc((size_t)NODES * 2 * HC * 4);
    _Float16* xp   = (_Float16*)P;   // alias: xp lifetime ends before P's begins

    const int T = 256;

    conv_wn_kernel<<<(512 * 128) / T, T, 0, stream>>>(Wn, WnT);
    conv_wa_kernel<<<(1024 * 512) / T, T, 0, stream>>>(Wa, BT);
    conv_x_kernel<<<(NODES * 128) / T, T, 0, stream>>>(x, xp);

    // h = x @ Wn + bn   (M=32768, N=512, K=128)
    {
        dim3 g(NODES / 128, 512 / 64);
        wmma_gemm_kernel<<<g, 256, 0, stream>>>(xp, WnT, bn, bufA, 512, 128);
    }

    float* hin = bufA;
    for (int layer = 0; layer < 2; ++layer) {
        float* hout = (layer == 0) ? bufB : out;

        conv_h_kernel<<<(NODES * HC) / T, T, 0, stream>>>(hin, hf16);

        // P[:,0:512] = h @ Wa_top ; P[:,512:1024] = h @ Wa_bot  (M=32768, N=1024, K=512)
        {
            dim3 g(NODES / 128, 1024 / 64);
            wmma_gemm_kernel<<<g, 256, 0, stream>>>(hf16, BT, nullptr, P, 1024, 512);
        }

        init_kernel<<<(NODES * HC) / T, T, 0, stream>>>(maxs, den, hout);

        int EH = EDGES * HEADS;
        score_kernel<<<EH / T, T, 0, stream>>>(P, src, dst, ba, attn, scr, maxs);
        expsum_kernel<<<EH / T, T, 0, stream>>>(dst, scr, maxs, den);
        aggregate_kernel<<<EH / T, T, 0, stream>>>(src, dst, scr, den, hin, hout);

        hin = hout;
    }
}
